// ExpertRouter_37237366456737
// MI455X (gfx1250) — matrix operand order repaired and checked by reading, not compiled, not run on hardware
//
#include <hip/hip_runtime.h>
#include <math.h>

// Problem constants (match reference)
#define B_ 4
#define L_ 2048
#define T_ (B_*L_)          // 8192 tokens
#define D_ 2048
#define E_ 8
#define K_ 2
#define H_ (4*D_)           // 8192
// FFN kernel tiling
#define HB 64               // hidden chunk per iteration
#define MT 32               // tokens per block (2 M-tiles of 16)
#define THREADS 512         // 16 waves
#define NPW 128             // output cols per wave (8 N-tiles)

typedef __attribute__((ext_vector_type(16))) __bf16 v16bf;
typedef __attribute__((ext_vector_type(2)))  __bf16 bf16x2;
typedef __attribute__((ext_vector_type(8)))  float  v8f;

// ---- f32 -> bf16 via hardware packed converter (v_cvt_pk_bf16_f32, RNE) ----
__device__ __forceinline__ bf16x2 pkbf(float a, float b) {
#if __has_builtin(__builtin_amdgcn_cvt_pk_bf16_f32)
  return __builtin_amdgcn_cvt_pk_bf16_f32(a, b);
#else
  bf16x2 r; r[0] = (__bf16)a; r[1] = (__bf16)b; return r;
#endif
}
__device__ __forceinline__ unsigned short f2bf_bits(float f) {
  return __builtin_bit_cast(unsigned short, (__bf16)f);
}

// Load two groups of 8 contiguous f32 and pack into a v16bf fragment
// (elements 0..7 from p0, 8..15 from p1). Pointers must be 16B aligned.
__device__ __forceinline__ v16bf cvt8x2(const float* p0, const float* p1) {
  const float4* f0 = (const float4*)p0;
  const float4* f1 = (const float4*)p1;
  float4 a0 = f0[0], a1 = f0[1], a2 = f1[0], a3 = f1[1];
  union { bf16x2 p[8]; v16bf v; } u;
  u.p[0] = pkbf(a0.x, a0.y);
  u.p[1] = pkbf(a0.z, a0.w);
  u.p[2] = pkbf(a1.x, a1.y);
  u.p[3] = pkbf(a1.z, a1.w);
  u.p[4] = pkbf(a2.x, a2.y);
  u.p[5] = pkbf(a2.z, a2.w);
  u.p[6] = pkbf(a3.x, a3.y);
  u.p[7] = pkbf(a3.z, a3.w);
  return u.v;
}

// A-fragment from bf16 LDS row: elements 0..7 at rowp+k0, 8..15 at rowp+k0+16
__device__ __forceinline__ v16bf afrag_lds(const unsigned short* rowp, int k0) {
  union { uint4 q[2]; v16bf v; } u;
  u.q[0] = *(const uint4*)(rowp + k0);
  u.q[1] = *(const uint4*)(rowp + k0 + 16);
  return u.v;
}

__device__ __forceinline__ v8f wmma_bf16(v16bf a, v16bf b, v8f c) {
  return __builtin_amdgcn_wmma_f32_16x16x32_bf16(false, a, false, b,
                                                 (short)0, c, false, false);
}

// ---------------- Kernel 0: zero output + expert counters ----------------
__global__ void moe_zero(float* __restrict__ out, long long ntotal,
                         int* __restrict__ cnt) {
  long long i = (long long)blockIdx.x * blockDim.x + threadIdx.x;
  if (i < ntotal) out[i] = 0.0f;
  if (i < E_) cnt[i] = 0;
}

// ---------------- Kernel 1: gating / top-2 routing ----------------
// One wave (32 lanes) per token.
__global__ void moe_route(const float* __restrict__ x,
                          const float* __restrict__ gw,
                          int* __restrict__ cnt, int* __restrict__ list,
                          float* __restrict__ combine) {
  const int wave = threadIdx.x >> 5;
  const int lane = threadIdx.x & 31;
  const int t = blockIdx.x * 8 + wave;
  if (t >= T_) return;

  float p[E_];
#pragma unroll
  for (int e = 0; e < E_; e++) p[e] = 0.0f;
  const float* xr = x + (size_t)t * D_;
  for (int d = lane; d < D_; d += 32) {
    float xv = xr[d];
#pragma unroll
    for (int e = 0; e < E_; e++) p[e] += xv * gw[e * D_ + d];
  }
#pragma unroll
  for (int off = 16; off > 0; off >>= 1) {
#pragma unroll
    for (int e = 0; e < E_; e++) p[e] += __shfl_down(p[e], off, 32);
  }
  if (lane == 0) {
    // top-2, ties -> lowest index first (jax.lax.top_k semantics)
    int e0 = 0; float m0 = p[0];
#pragma unroll
    for (int e = 1; e < E_; e++) if (p[e] > m0) { m0 = p[e]; e0 = e; }
    int e1 = -1; float m1 = -3.4e38f;
#pragma unroll
    for (int e = 0; e < E_; e++)
      if (e != e0 && p[e] > m1) { m1 = p[e]; e1 = e; }
    // softmax over the two selected logits (m1 <= m0)
    float z = __expf(m1 - m0);
    float g0 = 1.0f / (1.0f + z);
    float g1 = z / (1.0f + z);
    combine[t * E_ + e0] = g0;
    combine[t * E_ + e1] = g1;
    int p0 = atomicAdd(&cnt[e0], 1); list[e0 * T_ + p0] = t;
    int p1 = atomicAdd(&cnt[e1], 1); list[e1 * T_ + p1] = t;
  }
}

// ---------------- Kernel 2: fused expert FFN (WMMA bf16) ----------------
// grid = (T_/MT token-tiles, E_ experts); block = 512 threads (16 waves).
// Each block: 32 gathered tokens, full D output, H chunked by HB=64.
__global__ __launch_bounds__(THREADS, 1)
void moe_ffn(const float* __restrict__ x, const float* __restrict__ w1,
             const float* __restrict__ w2, const int* __restrict__ cnt,
             const int* __restrict__ list, const float* __restrict__ combine,
             float* __restrict__ out) {
  const int e = blockIdx.y;
  const int tile = blockIdx.x;
  const int n_e = cnt[e];
  if (tile * MT >= n_e) return;   // early-exit unused tiles (uniform)

  __shared__ int   s_tok[MT];
  __shared__ float s_gate[MT];
  __shared__ float s_hp[2][MT][HB];                       // K-split partials (16KB)
  __shared__ __align__(16) unsigned short s_hbf[MT][HB];  // gelu(h) in bf16 (4KB)

  const int tid = threadIdx.x;
  if (tid < MT) {
    int idx = tile * MT + tid;
    if (idx < n_e) {
      int t = list[e * T_ + idx];
      s_tok[tid] = t;
      s_gate[tid] = combine[t * E_ + e];
    } else {
      s_tok[tid] = 0;        // padded row: valid address, zero gate
      s_gate[tid] = 0.0f;
    }
  }
  __syncthreads();

  const int wave = tid >> 5;
  const int lane = tid & 31;
  const int lhalf = (lane < 16) ? 0 : 1;
  const int lmod = lane & 15;

  // GEMM1 role: q = K-half over D, mt1 = token tile, nt1 = hidden tile
  const int q   = wave >> 3;
  const int r   = wave & 7;
  const int mt1 = r >> 2;
  const int nt1 = r & 3;
  const int kb0 = q * (D_ / 2);
  const float* xrow = x + (size_t)s_tok[mt1 * 16 + lmod] * D_;
  const size_t w1_lane_off = (size_t)(nt1 * 16 + lmod) * D_ + (lhalf ? 8 : 0);
  const int a_off = lhalf ? 8 : 0;   // A-layout lane base (K += 8 for hi lanes)

  // GEMM2: this wave owns output columns [wnb, wnb+128)
  const int wnb = wave * NPW;

  v8f acc[16];
#pragma unroll
  for (int i = 0; i < 16; i++)
#pragma unroll
    for (int j = 0; j < 8; j++) acc[i][j] = 0.0f;

  for (int hb = 0; hb < H_; hb += HB) {
    // ---- GEMM1: h1_partial[16x16] = X[16 x 1024] * W1^T chunk ----
    v8f acc1;
#pragma unroll
    for (int j = 0; j < 8; j++) acc1[j] = 0.0f;
    const float* w1h = w1 + ((size_t)(e * H_ + hb)) * D_ + w1_lane_off;
    // prefetch next H-chunk of w1 (lowers to global_prefetch_b8)
    if (hb + HB < H_) __builtin_prefetch(w1h + (size_t)HB * D_, 0, 3);
#pragma unroll 4
    for (int ks = 0; ks < (D_ / 2) / 32; ks++) {
      int kb = kb0 + ks * 32;
      v16bf a = cvt8x2(xrow + kb + a_off, xrow + kb + a_off + 16);
      v16bf b = cvt8x2(w1h + kb, w1h + kb + 16);
      acc1 = wmma_bf16(a, b, acc1);
    }
#pragma unroll
    for (int j = 0; j < 8; j++)
      s_hp[q][mt1 * 16 + j + 8 * lhalf][nt1 * 16 + lmod] = acc1[j];
    __syncthreads();

    // ---- reduce K-halves + exact GELU + convert to bf16 ----
#pragma unroll
    for (int i = 0; i < (MT * HB) / THREADS; i++) {
      int idx = tid + i * THREADS;
      int rrow = idx >> 6, ccol = idx & 63;
      float v = s_hp[0][rrow][ccol] + s_hp[1][rrow][ccol];
      float g = 0.5f * v * (1.0f + erff(v * 0.70710678118f));
      s_hbf[rrow][ccol] = f2bf_bits(g);
    }
    __syncthreads();

    // ---- GEMM2: OUT[32 x 128] += gelu_tile[32 x 64] * W2^T chunk ----
#pragma unroll
    for (int ks2 = 0; ks2 < 2; ks2++) {
      const int k0 = ks2 * 32 + a_off;
      v16bf a0 = afrag_lds(&s_hbf[0 * 16 + lmod][0], k0);
      v16bf a1 = afrag_lds(&s_hbf[1 * 16 + lmod][0], k0);
#pragma unroll
      for (int nt = 0; nt < 8; nt++) {
        const float* w2p = w2 +
            ((size_t)(e * D_ + wnb + nt * 16 + lmod)) * H_ +
            hb + ks2 * 32 + (lhalf ? 8 : 0);
        v16bf b = cvt8x2(w2p, w2p + 16);
        acc[0 * 8 + nt] = wmma_bf16(a0, b, acc[0 * 8 + nt]);
        acc[1 * 8 + nt] = wmma_bf16(a1, b, acc[1 * 8 + nt]);
      }
    }
    // next iteration's s_hp writes are fenced by the first barrier above
  }

  // ---- epilogue: scale by gate, accumulate into output ----
#pragma unroll
  for (int mt = 0; mt < 2; mt++) {
#pragma unroll
    for (int j = 0; j < 8; j++) {
      int row = mt * 16 + j + 8 * lhalf;
      int t = s_tok[row];
      float g = s_gate[row];
      float* op = out + (size_t)t * D_ + wnb + lmod;
#pragma unroll
      for (int nt = 0; nt < 8; nt++)
        atomicAdd(op + nt * 16, g * acc[mt * 8 + nt][j]);
    }
  }
}

// ---------------- Kernel 3: load-balance loss ----------------
__global__ void moe_loss(const int* __restrict__ cnt, float* __restrict__ out,
                         int loss_idx) {
  if (blockIdx.x == 0 && threadIdx.x == 0) {
    float tot = 0.0f, c[E_];
#pragma unroll
    for (int e = 0; e < E_; e++) { c[e] = (float)cnt[e]; tot += c[e]; }
    float l = 0.0f;
#pragma unroll
    for (int e = 0; e < E_; e++) {
      float d = c[e] / tot - 1.0f / (float)E_;
      l += d * d;
    }
    out[loss_idx] = l / (float)E_;
  }
}

extern "C" void kernel_launch(void* const* d_in, const int* in_sizes, int n_in,
                              void* d_out, int out_size, void* d_ws, size_t ws_size,
                              hipStream_t stream) {
  const float* x  = (const float*)d_in[0];
  const float* gw = (const float*)d_in[1];
  const float* w1 = (const float*)d_in[2];
  const float* w2 = (const float*)d_in[3];
  float* out = (float*)d_out;

  // workspace layout (~513KB): counters | token lists | combine gates
  char* ws = (char*)d_ws;
  int*   cnt     = (int*)ws;
  int*   list    = (int*)(ws + 64);
  float* combine = (float*)(ws + 64 + (size_t)E_ * T_ * sizeof(int));

  const long long ntotal = (long long)T_ * D_;
  const int zb = (int)((ntotal + 255) / 256);
  moe_zero<<<zb, 256, 0, stream>>>(out, ntotal, cnt);
  moe_route<<<T_ / 8, 256, 0, stream>>>(x, gw, cnt, list, combine);
  dim3 grid(T_ / MT, E_);
  moe_ffn<<<grid, THREADS, 0, stream>>>(x, w1, w2, cnt, list, combine, out);
  moe_loss<<<1, 32, 0, stream>>>(cnt, out, out_size - 1);
}
